// TDSDecoder_79499844649548
// MI455X (gfx1250) — matrix operand order repaired and checked by reading, not compiled, not run on hardware
//
#include <hip/hip_runtime.h>

#define VOCAB 40
#define EDIM 512
#define HDIM 512
#define DKD 512
#define DVD 512
#define TENC 8192
#define MAX_LEN 200
#define EOS_IDX 38
#define NTHREADS 512
#define NWAVES 16

typedef _Float16 v16h __attribute__((ext_vector_type(16)));
typedef float    v8f  __attribute__((ext_vector_type(8)));

union Frag16 { v16h v; uint4 q[2]; };

__device__ __forceinline__ float waveReduceSum(float v) {
  #pragma unroll
  for (int o = 16; o > 0; o >>= 1) v += __shfl_down(v, o, 32);
  return v;
}

// Prologue 1: keys -> f16 row-major; values -> f16 transposed (512 x 8192).
__global__ void tds_cvt_kernel(const float* __restrict__ enc,
                               _Float16* __restrict__ keys16,
                               _Float16* __restrict__ valt16) {
  int idx = blockIdx.x * blockDim.x + threadIdx.x;   // 8192*512 threads
  int r = idx >> 9;          // 0..8191 (time)
  int c = idx & 511;         // 0..511  (feature)
  keys16[(size_t)r * DKD + c]  = (_Float16)enc[(size_t)r * (DKD + DVD) + c];
  valt16[(size_t)c * TENC + r] = (_Float16)enc[(size_t)r * (DKD + DVD) + DKD + c];
}

// Prologue 2: GRU weights -> f16 row-major.
__global__ void tds_cvt_w_kernel(const float* __restrict__ w_ih,
                                 const float* __restrict__ w_hh,
                                 _Float16* __restrict__ wih16,
                                 _Float16* __restrict__ whh16) {
  int idx = blockIdx.x * blockDim.x + threadIdx.x;   // 1536*512 threads
  wih16[idx] = (_Float16)w_ih[idx];
  whh16[idx] = (_Float16)w_hh[idx];
}

__global__ __launch_bounds__(NTHREADS, 1)
void tds_decoder_kernel(const float* __restrict__ hidden,
                        const float* __restrict__ embed,
                        const float* __restrict__ b_ih,
                        const float* __restrict__ b_hh,
                        const float* __restrict__ w_out,
                        const float* __restrict__ b_out,
                        const _Float16* __restrict__ keys16,
                        const _Float16* __restrict__ valt16,
                        const _Float16* __restrict__ wih16,
                        const _Float16* __restrict__ whh16,
                        float* __restrict__ out) {
  __shared__ float s_h[HDIM];                        // f32 h (output-layer concat)
  __shared__ __align__(16) _Float16 s_h16[HDIM];     // f16 h (WMMA B)
  __shared__ __align__(16) _Float16 s_x16[EDIM];     // f16 embed[best] (WMMA B)
  __shared__ float s_summary[DVD];
  __shared__ __align__(16) _Float16 s_vec16[TENC];   // f16 scores for summary WMMA
  __shared__ float s_big[TENC];                      // gi/gh, then logits/probs
  __shared__ float s_red[NWAVES];
  __shared__ float s_out[VOCAB];
  __shared__ float s_bcast;
  __shared__ int   s_best;
  __shared__ int   s_eos;

  const int tid  = threadIdx.x;
  const int lane = tid & 31;
  const int wave = tid >> 5;
  const float scale = 0.0441941738241592f;  // 1/sqrt(512)

  for (int i = tid; i < HDIM; i += NTHREADS) {
    float h0 = hidden[i];
    s_h[i]   = h0;
    s_h16[i] = (_Float16)h0;
    s_x16[i] = (_Float16)embed[(size_t)EOS_IDX * EDIM + i];
  }
  if (tid == 0) s_eos = MAX_LEN;
  __syncthreads();

  // Per-lane WMMA fragment geometry (16-bit A 16x32 / B 32x16 layouts).
  const int mrow = lane & 15;
  const int o1   = (lane < 16) ? 0 : 16;  // A-frag: K{0..15} vs {16..31}
  const int boff = (lane < 16) ? 0 : 16;  // B-frag: K{0..15} vs {16..31}

  for (int step = 0; step < MAX_LEN; ++step) {
    // ---------- GRU gate matvecs via WMMA: gi = w_ih@x, gh = w_hh@h ----------
    // 96 row-tiles per matrix; each wave owns 6, processed as 3 pairs.
    for (int j = 0; j < 3; ++j) {
      const int t0 = wave + 32 * j;
      const int t1 = t0 + 16;
      {  // gi
        const _Float16* arow0 = wih16 + (size_t)(t0 * 16 + mrow) * EDIM;
        const _Float16* arow1 = wih16 + (size_t)(t1 * 16 + mrow) * EDIM;
        v8f acc0 = {}, acc1 = {};
        #pragma unroll 2
        for (int kb = 0; kb < EDIM; kb += 32) {
          Frag16 a0, a1, b;
          b.q[0]  = *(const uint4*)(&s_x16[kb + boff]);
          b.q[1]  = *(const uint4*)(&s_x16[kb + boff + 8]);
          a0.q[0] = *(const uint4*)(arow0 + kb + o1);
          a0.q[1] = *(const uint4*)(arow0 + kb + o1 + 8);
          a1.q[0] = *(const uint4*)(arow1 + kb + o1);
          a1.q[1] = *(const uint4*)(arow1 + kb + o1 + 8);
          acc0 = __builtin_amdgcn_wmma_f32_16x16x32_f16(
              false, a0.v, false, b.v, (short)0, acc0, false, false);
          acc1 = __builtin_amdgcn_wmma_f32_16x16x32_f16(
              false, a1.v, false, b.v, (short)0, acc1, false, false);
        }
        if (lane == 0) {
          #pragma unroll
          for (int r = 0; r < 8; ++r) {
            s_big[t0 * 16 + r] = acc0[r];
            s_big[t1 * 16 + r] = acc1[r];
          }
        }
        if (lane == 16) {
          #pragma unroll
          for (int r = 0; r < 8; ++r) {
            s_big[t0 * 16 + 8 + r] = acc0[r];
            s_big[t1 * 16 + 8 + r] = acc1[r];
          }
        }
      }
      {  // gh
        const _Float16* arow0 = whh16 + (size_t)(t0 * 16 + mrow) * HDIM;
        const _Float16* arow1 = whh16 + (size_t)(t1 * 16 + mrow) * HDIM;
        v8f acc0 = {}, acc1 = {};
        #pragma unroll 2
        for (int kb = 0; kb < HDIM; kb += 32) {
          Frag16 a0, a1, b;
          b.q[0]  = *(const uint4*)(&s_h16[kb + boff]);
          b.q[1]  = *(const uint4*)(&s_h16[kb + boff + 8]);
          a0.q[0] = *(const uint4*)(arow0 + kb + o1);
          a0.q[1] = *(const uint4*)(arow0 + kb + o1 + 8);
          a1.q[0] = *(const uint4*)(arow1 + kb + o1);
          a1.q[1] = *(const uint4*)(arow1 + kb + o1 + 8);
          acc0 = __builtin_amdgcn_wmma_f32_16x16x32_f16(
              false, a0.v, false, b.v, (short)0, acc0, false, false);
          acc1 = __builtin_amdgcn_wmma_f32_16x16x32_f16(
              false, a1.v, false, b.v, (short)0, acc1, false, false);
        }
        if (lane == 0) {
          #pragma unroll
          for (int r = 0; r < 8; ++r) {
            s_big[1536 + t0 * 16 + r] = acc0[r];
            s_big[1536 + t1 * 16 + r] = acc1[r];
          }
        }
        if (lane == 16) {
          #pragma unroll
          for (int r = 0; r < 8; ++r) {
            s_big[1536 + t0 * 16 + 8 + r] = acc0[r];
            s_big[1536 + t1 * 16 + 8 + r] = acc1[r];
          }
        }
      }
    }
    __syncthreads();

    // ---------- gate combine (biases added here, coalesced) ----------
    {
      float gir = s_big[tid]            + b_ih[tid];
      float giz = s_big[HDIM + tid]     + b_ih[HDIM + tid];
      float gin = s_big[2 * HDIM + tid] + b_ih[2 * HDIM + tid];
      float ghr = s_big[1536 + tid]            + b_hh[tid];
      float ghz = s_big[1536 + HDIM + tid]     + b_hh[HDIM + tid];
      float ghn = s_big[1536 + 2 * HDIM + tid] + b_hh[2 * HDIM + tid];
      float r = 1.f / (1.f + expf(-(gir + ghr)));
      float z = 1.f / (1.f + expf(-(giz + ghz)));
      float n = tanhf(gin + r * ghn);
      float hnew = (1.f - z) * n + z * s_h[tid];
      s_h[tid]   = hnew;
      s_h16[tid] = (_Float16)hnew;
    }
    __syncthreads();

    // ---------- attention logits: keys16 (8192x512) @ h via WMMA f16 ----------
    for (int j = 0; j < 16; ++j) {
      const int t0 = wave + 32 * j;
      const int t1 = t0 + 16;
      const _Float16* arow0 = keys16 + (size_t)(t0 * 16 + mrow) * DKD;
      const _Float16* arow1 = keys16 + (size_t)(t1 * 16 + mrow) * DKD;
      if (j < 15) {  // warm next pair's rows while we compute this pair
        __builtin_prefetch(arow0 + 32 * 16 * DKD, 0, 3);
        __builtin_prefetch(arow1 + 32 * 16 * DKD, 0, 3);
      }
      v8f acc0 = {}, acc1 = {};
      #pragma unroll 2
      for (int kb = 0; kb < DKD; kb += 32) {
        Frag16 a0, a1, b;
        b.q[0]  = *(const uint4*)(&s_h16[kb + boff]);
        b.q[1]  = *(const uint4*)(&s_h16[kb + boff + 8]);
        a0.q[0] = *(const uint4*)(arow0 + kb + o1);
        a0.q[1] = *(const uint4*)(arow0 + kb + o1 + 8);
        a1.q[0] = *(const uint4*)(arow1 + kb + o1);
        a1.q[1] = *(const uint4*)(arow1 + kb + o1 + 8);
        acc0 = __builtin_amdgcn_wmma_f32_16x16x32_f16(
            false, a0.v, false, b.v, (short)0, acc0, false, false);
        acc1 = __builtin_amdgcn_wmma_f32_16x16x32_f16(
            false, a1.v, false, b.v, (short)0, acc1, false, false);
      }
      if (lane == 0) {
        #pragma unroll
        for (int r = 0; r < 8; ++r) {
          s_big[t0 * 16 + r] = acc0[r] * scale;
          s_big[t1 * 16 + r] = acc1[r] * scale;
        }
      }
      if (lane == 16) {
        #pragma unroll
        for (int r = 0; r < 8; ++r) {
          s_big[t0 * 16 + 8 + r] = acc0[r] * scale;
          s_big[t1 * 16 + 8 + r] = acc1[r] * scale;
        }
      }
    }
    __syncthreads();

    // ---------- softmax over 8192 logits ----------
    float lm = -3.4e38f;
    for (int i = tid; i < TENC; i += NTHREADS) lm = fmaxf(lm, s_big[i]);
    #pragma unroll
    for (int o = 16; o > 0; o >>= 1) lm = fmaxf(lm, __shfl_down(lm, o, 32));
    if (lane == 0) s_red[wave] = lm;
    __syncthreads();
    if (tid == 0) {
      float m = s_red[0];
      for (int w = 1; w < NWAVES; ++w) m = fmaxf(m, s_red[w]);
      s_bcast = m;
    }
    __syncthreads();
    const float mmax = s_bcast;
    float ls = 0.f;
    for (int i = tid; i < TENC; i += NTHREADS) {
      float e = expf(s_big[i] - mmax);
      s_big[i] = e;
      ls += e;
    }
    ls = waveReduceSum(ls);
    if (lane == 0) s_red[wave] = ls;
    __syncthreads();
    if (tid == 0) {
      float s = 0.f;
      for (int w = 0; w < NWAVES; ++w) s += s_red[w];
      s_bcast = s;
    }
    __syncthreads();
    const float inv = 1.f / s_bcast;
    float* attn = out + 8001 + (size_t)step * TENC;
    for (int i = tid; i < TENC; i += NTHREADS) {
      float p = s_big[i] * inv;
      attn[i] = p;                 // attentions output
      s_vec16[i] = (_Float16)p;    // scores in f16 for summary WMMA
    }
    __syncthreads();

    // ---------- summary = values^T (512x8192) @ scores via WMMA f16 ----------
    {
      const _Float16* arow0 = valt16 + (size_t)(wave * 16 + mrow) * TENC;
      const _Float16* arow1 = valt16 + (size_t)((wave + 16) * 16 + mrow) * TENC;
      v8f acc0 = {}, acc1 = {};
      #pragma unroll 2
      for (int kb = 0; kb < TENC; kb += 32) {
        Frag16 a0, a1, b;
        b.q[0]  = *(const uint4*)(&s_vec16[kb + boff]);
        b.q[1]  = *(const uint4*)(&s_vec16[kb + boff + 8]);
        a0.q[0] = *(const uint4*)(arow0 + kb + o1);
        a0.q[1] = *(const uint4*)(arow0 + kb + o1 + 8);
        a1.q[0] = *(const uint4*)(arow1 + kb + o1);
        a1.q[1] = *(const uint4*)(arow1 + kb + o1 + 8);
        acc0 = __builtin_amdgcn_wmma_f32_16x16x32_f16(
            false, a0.v, false, b.v, (short)0, acc0, false, false);
        acc1 = __builtin_amdgcn_wmma_f32_16x16x32_f16(
            false, a1.v, false, b.v, (short)0, acc1, false, false);
      }
      if (lane == 0) {
        #pragma unroll
        for (int r = 0; r < 8; ++r) {
          s_summary[wave * 16 + r]        = acc0[r];
          s_summary[(wave + 16) * 16 + r] = acc1[r];
        }
      }
      if (lane == 16) {
        #pragma unroll
        for (int r = 0; r < 8; ++r) {
          s_summary[wave * 16 + 8 + r]        = acc0[r];
          s_summary[(wave + 16) * 16 + 8 + r] = acc1[r];
        }
      }
    }
    __syncthreads();

    // ---------- output layer: 40 x 1024 matvec (f32) ----------
    for (int row = wave; row < VOCAB; row += NWAVES) {
      const float* wr = w_out + (size_t)row * (DVD + HDIM);
      float acc = 0.f;
      for (int k = lane; k < DVD + HDIM; k += 32) {
        float xv = (k < DVD) ? s_summary[k] : s_h[k - DVD];
        acc += wr[k] * xv;
      }
      acc = waveReduceSum(acc);
      if (lane == 0) s_out[row] = acc + b_out[row];
    }
    __syncthreads();

    // ---------- argmax + EOS tracking ----------
    if (tid == 0) {
      int best = 0;
      float bv = s_out[0];
      for (int v = 1; v < VOCAB; ++v) {
        float x = s_out[v];
        if (x > bv) { bv = x; best = v; }
      }
      s_best = best;
      if (best == EOS_IDX && s_eos == MAX_LEN) s_eos = step;
    }
    __syncthreads();

    if (tid < VOCAB) out[(size_t)step * VOCAB + tid] = s_out[tid];
    const int best = s_best;
    for (int i = tid; i < EDIM; i += NTHREADS)
      s_x16[i] = (_Float16)embed[(size_t)best * EDIM + i];
    __syncthreads();
  }

  if (tid == 0) ((int*)out)[8000] = s_eos;  // lens (int32)
}

extern "C" void kernel_launch(void* const* d_in, const int* in_sizes, int n_in,
                              void* d_out, int out_size, void* d_ws, size_t ws_size,
                              hipStream_t stream) {
  (void)in_sizes; (void)n_in; (void)out_size; (void)ws_size;
  const float* enc    = (const float*)d_in[0];
  const float* hidden = (const float*)d_in[1];
  const float* embed  = (const float*)d_in[2];
  const float* w_ih   = (const float*)d_in[3];
  const float* w_hh   = (const float*)d_in[4];
  const float* b_ih   = (const float*)d_in[5];
  const float* b_hh   = (const float*)d_in[6];
  const float* w_out  = (const float*)d_in[7];
  const float* b_out  = (const float*)d_in[8];
  float* out = (float*)d_out;

  char* ws = (char*)d_ws;
  _Float16* keys16 = (_Float16*)(ws);                                   // 8 MB
  _Float16* valt16 = (_Float16*)(ws + (size_t)TENC * DKD * 2);          // 8 MB
  _Float16* wih16  = (_Float16*)(ws + (size_t)TENC * DKD * 4);          // 1.5 MB
  _Float16* whh16  = (_Float16*)(ws + (size_t)TENC * DKD * 4
                                    + (size_t)3 * HDIM * EDIM * 2);     // 1.5 MB

  tds_cvt_kernel<<<(TENC * DKD) / 256, 256, 0, stream>>>(enc, keys16, valt16);
  tds_cvt_w_kernel<<<(3 * HDIM * EDIM) / 256, 256, 0, stream>>>(w_ih, w_hh,
                                                                wih16, whh16);
  tds_decoder_kernel<<<1, NTHREADS, 0, stream>>>(hidden, embed, b_ih, b_hh,
                                                 w_out, b_out,
                                                 keys16, valt16, wih16, whh16,
                                                 out);
}
